// CgpHmmCell_70291434766847
// MI455X (gfx1250) — matrix operand order repaired and checked by reading, not compiled, not run on hardware
//
#include <hip/hip_runtime.h>
#include <cstdint>

// ---------------- problem constants ----------------
#define NSTATES 612      // 6*100+12
#define SA      33       // padded LDS row stride (gcd(33,64)=1 -> conflict-free)
#define SPE     640      // padded row stride for global A / Et
#define NB      32       // batch
#define TLEN    2048
#define NEDGE   5866
#define NEMIT   216

typedef __attribute__((ext_vector_type(16))) __bf16 v16bf;
typedef __attribute__((ext_vector_type(8)))  float  v8f;

// ---------------- static sparse structure of the transition kernel ----------
// Decodes edge e -> (row, col, raw value) exactly mirroring the reference's
// get_indices_and_values ordering (order only matters for (idx,val) pairing,
// which we do jointly per group).
__device__ inline void edge_decode(int e, const float* w, int& r, int& c, float& v) {
  if (e == 0)        { r = 0; c = 0; v = 1.f - w[0]; }
  else if (e == 1)   { r = 0; c = 1; v = w[0]; }
  else if (e == 2)   { r = 1; c = 2; v = 1.f; }
  else if (e == 3)   { r = 2; c = 3; v = 1.f; }
  else if (e < 104)  { int i = e - 4;   r = 3 + 3*i; c = 4 + 3*i;   v = w[1 + i]; }
  else if (e < 204)  { int i = e - 104; r = 4 + 3*i; c = 5 + 3*i;   v = 1.f; }
  else if (e < 304)  { int i = e - 204; r = 5 + 3*i; c = 6 + 3*i;   v = 1.f; }
  else if (e < 405)  { int i = e - 304; r = 3 + 3*i; c = 308 + 3*i; v = w[101 + i]; }
  else if (e < 406)  { r = 303; c = 304; v = w[202]; }
  else if (e < 507)  { int i = e - 406; r = 308 + 3*i; c = 309 + 3*i; v = 1.f; }
  else if (e < 608)  { int i = e - 507; r = 309 + 3*i; c = 310 + 3*i; v = 1.f; }
  else if (e < 709)  { int i = e - 608; r = 310 + 3*i; c = 4 + 3*i;   v = w[203 + i]; }
  else if (e < 810)  { int i = e - 709; r = 310 + 3*i; c = 308 + 3*i; v = 1.f - w[203 + i]; }
  else if (e < 5860) {                       // delete transitions (triangular block)
    int d = e - 810;
    int i = 0, rem = 100;
    while (d >= rem) { d -= rem; --rem; ++i; }
    int j = i + 1 + d;
    r = 3 + 3*i; c = 4 + 3*j;
    float wk = w[304];
    float p = wk;                            // wk^(j-i+1), integer power (sign-correct)
    for (int t = 0; t < j - i; ++t) p *= wk;
    v = 1.f - p;
  } else {
    int k = e - 5860;
    const int rr[6] = {304, 305, 306, 307, 307, 611};
    const int cc[6] = {305, 306, 307, 307, 611, 611};
    r = rr[k]; c = cc[k]; v = 1.f;
  }
}

// ---------------- one-shot preprocessing kernels ----------------
__global__ void k_zero(float* p, int n) {
  int i = blockIdx.x * blockDim.x + threadIdx.x;
  if (i < n) p[i] = 0.f;
}

__global__ void k_edge_accum(const float* __restrict__ w, float* __restrict__ rowsum) {
  int e = blockIdx.x * blockDim.x + threadIdx.x;
  if (e >= NEDGE) return;
  int r, c; float v;
  edge_decode(e, w, r, c, v);
  atomicAdd(&rowsum[r], expf(v));
}

__global__ void k_edge_write(const float* __restrict__ w, const float* __restrict__ rowsum,
                             float* __restrict__ Ad) {
  int e = blockIdx.x * blockDim.x + threadIdx.x;
  if (e >= NEDGE) return;
  int r, c; float v;
  edge_decode(e, w, r, c, v);
  Ad[r * SPE + c] = expf(v) / rowsum[r];     // row-wise softmax over structural entries
}

__global__ void k_emit(const float* __restrict__ ek, float* __restrict__ Et) {
  int r = blockIdx.x * blockDim.x + threadIdx.x;
  if (r >= NSTATES) return;
  float sum = 0.f;
  for (int o = 0; o < NEMIT; ++o) sum += expf(ek[r * NEMIT + o]);
  float inv = 1.f / sum;
  for (int o = 0; o < NEMIT; ++o) Et[o * SPE + r] = expf(ek[r * NEMIT + o]) * inv;
}

// ---------------- WMMA fragment helpers (CDNA5 16x16x32 bf16 layouts) -------
// A-matrix 16x32 bf16: lane = M row (0-15 twice); lanes<16 hold K base+0..7 &
// base+16..23, lanes>=16 hold K base+8..15 & base+24..31 (pairs per VGPR).
__device__ inline v16bf load_afrag(const __bf16* mbf, int mt, int kt, int lane) {
  int m  = mt * 16 + (lane & 15);
  int k0 = kt * 32 + ((lane < 16) ? 0 : 8);
  const uint32_t* p0 = (const uint32_t*)(mbf + m * 128 + k0);
  const uint32_t* p1 = (const uint32_t*)(mbf + m * 128 + k0 + 16);
  union { v16bf v; uint32_t u[8]; } r;
#pragma unroll
  for (int i = 0; i < 4; ++i) { r.u[i] = p0[i]; r.u[4 + i] = p1[i]; }
  return r.v;
}

// ---------------- persistent forward-scan kernel ----------------
// Single workgroup, 16 wave32s. LDS-resident alpha/anew; delete-block B-matrix
// resident in VGPRs (waves 0..6); 56 v_wmma_f32_16x16x32_bf16 per time step.
__global__ void __launch_bounds__(512, 1)
k_forward(const int* __restrict__ ids, const float* __restrict__ ik,
          const float* __restrict__ Ad, const float* __restrict__ Et,
          float* __restrict__ out) {
  extern __shared__ char smem[];
  float*   alpha = (float*)smem;                       // [612][33]
  float*   anew  = alpha + NSTATES * SA;               // [612][33]
  __bf16*  mbf   = (__bf16*)(anew + NSTATES * SA);     // [32][128] compact bf16 alpha
  float*   v0t   = (float*)(mbf + 32 * 128);           // per-column sparse tables
  float*   v1t   = v0t + NSTATES;
  int*     s0t   = (int*)(v1t + NSTATES);
  int*     s1t   = s0t + NSTATES;
  float*   ll    = (float*)(s1t + NSTATES);            // [32]
  float*   red   = ll + NB;                            // [1]

  const int tid  = threadIdx.x;
  const int wid  = tid >> 5;
  const int lane = tid & 31;

  // ---- init: zero pads, ll, pi-softmax into anew ----
  for (int i = tid; i < 32 * 128; i += 512) mbf[i] = (__bf16)0.f;
  if (tid < NB) ll[tid] = 0.f;
  if (tid == 0) red[0] = 0.f;
  __syncthreads();

  float ps = 0.f;
  for (int s = tid; s < NSTATES; s += 512) ps += expf(ik[s]);
#pragma unroll
  for (int m = 16; m >= 1; m >>= 1) ps += __shfl_xor(ps, m, 32);
  if (lane == 0) atomicAdd(red, ps);
  __syncthreads();
  float rinv = 1.f / red[0];
  for (int s = wid; s < NSTATES; s += 16) {
    float pv = expf(ik[s]) * rinv;
    anew[s * SA + lane] = pv;                // a_pre(t=0) = pi, broadcast over batch
  }

  // ---- build sparse per-column tables (<=2 in-edges outside dense block) ----
  for (int c = tid; c < NSTATES; c += 512) {
    int a = -1, b2 = -1;
    if      (c == 0)   { a = 0; }
    else if (c == 1)   { a = 0; }
    else if (c == 2)   { a = 1; }
    else if (c == 3)   { a = 2; }
    else if (c == 305) { a = 304; }
    else if (c == 307) { a = 306; b2 = 307; }
    else if (c == 611) { a = 307; b2 = 611; }
    else if (c >= 308 && c <= 610) {
      int r = c - 308, i = r / 3, m = r % 3;
      if (m == 0)      { a = 3 + 3*i; b2 = 310 + 3*i; }
      else if (m == 1) { a = 308 + 3*i; }
      else             { a = 309 + 3*i; }
    } else if (c >= 4) {
      int r = c - 4, i = r / 3, m = r % 3;
      if (m == 0)      { a = 310 + 3*i; }            // dense block covers 3+3k rows
      else if (m == 1) { if (i <= 99) a = 4 + 3*i; }
      else             { a = (i <= 99) ? 5 + 3*i : 305; }
    }
    s0t[c] = (a  >= 0) ? a  : 0;  v0t[c] = (a  >= 0) ? Ad[a  * SPE + c] : 0.f;
    s1t[c] = (b2 >= 0) ? b2 : 0;  v1t[c] = (b2 >= 0) ? Ad[b2 * SPE + c] : 0.f;
  }

  // ---- load resident delete-block B fragments (waves 0..6) ----
  // B-matrix 32x16 bf16: lane = N col (0-15 twice); lanes<16 K=kb+0..7 & kb+16..23,
  // lanes>=16 K=kb+8..15 & kb+24..31 (same order as A fragment).
  v16bf dfrag[4];
  const int ng = wid * 16 + (lane & 15);
  if (wid < 7) {
#pragma unroll
    for (int kt = 0; kt < 4; ++kt) {
      union { v16bf v; uint32_t u[8]; } r;
      int kb = kt * 32 + ((lane < 16) ? 0 : 8);
#pragma unroll
      for (int p = 0; p < 8; ++p) {
        int k0 = kb + 2 * p + ((p >= 4) ? 8 : 0), k1 = k0 + 1;
        float f0 = (k0 <= 100 && ng <= 100) ? Ad[(3 + 3 * k0) * SPE + (4 + 3 * ng)] : 0.f;
        float f1 = (k1 <= 100 && ng <= 100) ? Ad[(3 + 3 * k1) * SPE + (4 + 3 * ng)] : 0.f;
        union { __bf16 h[2]; uint32_t u; } pk;
        pk.h[0] = (__bf16)f0; pk.h[1] = (__bf16)f1;
        r.u[p] = pk.u;
      }
      dfrag[kt] = r.v;
    }
  }
  __syncthreads();

  // ---- the serial scan ----
  for (int t = 0; t < TLEN; ++t) {
    // stage 3: alpha = normalize(anew * Et[ob_t]); ll += log(sum)
    // each wave owns 2 batch rows entirely -> wave-local reduction only
#pragma unroll
    for (int which = 0; which < 2; ++which) {
      int b  = wid * 2 + which;
      int ob = ids[b * TLEN + t];
      const float* Erow = Et + ob * SPE;
      float sum = 0.f;
      for (int s = lane; s < NSTATES; s += 32) {
        float p = anew[s * SA + b] * Erow[s];
        alpha[s * SA + b] = p;
        sum += p;
      }
#pragma unroll
      for (int m = 16; m >= 1; m >>= 1) sum += __shfl_xor(sum, m, 32);
      float inv = 1.f / sum;
      for (int s = lane; s < NSTATES; s += 32) alpha[s * SA + b] *= inv;
      if (lane == 0) ll[b] += logf(sum);
    }
    __syncthreads();

    if (t + 1 < TLEN) {
      // phase A: sparse columns (lane = batch), plus compact bf16 gather of
      // the match-state rows for the WMMA A-matrix (waves 8..15)
      for (int c = wid; c < NSTATES; c += 16) {
        float va = v0t[c], vb = v1t[c];
        int   sa = s0t[c], sb = s1t[c];
        anew[c * SA + lane] = va * alpha[sa * SA + lane] + vb * alpha[sb * SA + lane];
      }
      if (tid >= 256) {
        int tid2 = tid - 256;
        for (int idx = tid2; idx < 101 * 32; idx += 256) {
          int k = idx >> 5, b = idx & 31;
          mbf[b * 128 + k] = (__bf16)alpha[(3 + 3 * k) * SA + b];
        }
      }
      __syncthreads();

      // phase B: dense delete-block via WMMA (waves 0..6), scatter-add to anew
      if (wid < 7) {
        v8f acc0 = {}, acc1 = {};
#pragma unroll
        for (int kt = 0; kt < 4; ++kt) {
          v16bf a0 = load_afrag(mbf, 0, kt, lane);
          v16bf a1 = load_afrag(mbf, 1, kt, lane);
          acc0 = __builtin_amdgcn_wmma_f32_16x16x32_bf16(false, a0, false, dfrag[kt],
                                                         (short)0, acc0, false, false);
          acc1 = __builtin_amdgcn_wmma_f32_16x16x32_bf16(false, a1, false, dfrag[kt],
                                                         (short)0, acc1, false, false);
        }
        int col  = 4 + 3 * ng;                 // <= 337 < 612
        int boff = (lane < 16) ? 0 : 8;        // C layout: VGPR v -> M = v or 8+v
#pragma unroll
        for (int v = 0; v < 8; ++v) {
          atomicAdd(&anew[col * SA + (boff + v)],       acc0[v]);   // batches 0..15
          atomicAdd(&anew[col * SA + 16 + (boff + v)],  acc1[v]);   // batches 16..31
        }
      }
      __syncthreads();
    }
  }

  __syncthreads();
  if (tid < NB) out[tid] = ll[tid];
}

// ---------------- launch ----------------
extern "C" void kernel_launch(void* const* d_in, const int* in_sizes, int n_in,
                              void* d_out, int out_size, void* d_ws, size_t ws_size,
                              hipStream_t stream) {
  (void)in_sizes; (void)n_in; (void)out_size; (void)ws_size;
  const int*   ids = (const int*)d_in[0];
  const float* tk  = (const float*)d_in[1];
  const float* ek  = (const float*)d_in[2];
  const float* ik  = (const float*)d_in[3];
  float*       out = (float*)d_out;

  float* Ad     = (float*)d_ws;            // [612][640] dense softmaxed A (~1.5MB)
  float* rowsum = Ad + NSTATES * SPE;      // [640]
  float* Et     = rowsum + SPE;            // [216][640] transposed emissions (~0.55MB)

  int nz = NSTATES * SPE + SPE;
  k_zero<<<(nz + 255) / 256, 256, 0, stream>>>(Ad, nz);
  k_edge_accum<<<(NEDGE + 255) / 256, 256, 0, stream>>>(tk, rowsum);
  k_edge_write<<<(NEDGE + 255) / 256, 256, 0, stream>>>(tk, rowsum, Ad);
  k_emit<<<(NSTATES + 255) / 256, 256, 0, stream>>>(ek, Et);

  size_t smem = (size_t)(NSTATES * SA * 2) * 4      // alpha + anew
              + (size_t)(32 * 128) * 2              // mbf
              + (size_t)(NSTATES * 4 + NB + 1) * 4; // tables + ll + red
  smem = (smem + 255) & ~(size_t)255;
  hipFuncSetAttribute((const void*)k_forward,
                      hipFuncAttributeMaxDynamicSharedMemorySize, (int)smem);
  k_forward<<<1, 512, smem, stream>>>(ids, ik, Ad, Et, out);
}
